// DAWN_85899345920732
// MI455X (gfx1250) — matrix operand order repaired and checked by reading, not compiled, not run on hardware
//
#include <hip/hip_runtime.h>
#include <cstdint>
#include <cstddef>

// ---------------- problem sizes ----------------
constexpr int Bq  = 4;
constexpr int Sq  = 2048;
constexpr int Dd  = 1024;
constexpr int Nn  = 32768;
constexpr int NCH = 8;
constexpr int Mm  = Bq * Sq;    // 8192 tokens
constexpr int CS  = Nn / NCH;   // 4096 neurons per chunk

constexpr int LDK = 40;   // 32 + 8 pad (bf16 elems) -> conflict-free row access
constexpr int LDW = 136;  // 128 + 8 pad

// row-stat slots in workspace
enum { RS_SSUM = 0, RS_SQ, RS_CUBE, RS_TAU, RS_STD, RS_MEAN, RS_WCOST, RS_GMAX,
       RS_ACT, RS_STRONG, RS_PHIB, RS_ZSUM, RS_Z075, RS_Z030, RS_GLG, RS_COUNT };
// scalar accumulator slots
enum { SC_SKEW = 0, SC_STDSUM, SC_NSSUM, SC_NSSQ, SC_ES, SC_ACT, SC_ACTSQ,
       SC_Z075, SC_Z030, SC_TAU, SC_ENT, SC_COUNT };

// ---------------- workspace layout (bytes) ----------------
constexpr size_t WS_EMB   = 0;                                   // [N,D] bf16 (normalized)
constexpr size_t WS_READ  = WS_EMB   + (size_t)Nn * Dd * 2;      // [N,D] bf16 (normalized)
constexpr size_t WS_WRITE = WS_READ  + (size_t)Nn * Dd * 2;      // [N,D] bf16 (normalized)
constexpr size_t WS_HBF   = WS_WRITE + (size_t)Nn * Dd * 2;      // [M,D] bf16
constexpr size_t WS_XBF   = WS_HBF   + (size_t)Mm * Dd * 2;      // [M,D] bf16
constexpr size_t WS_GXR   = WS_XBF   + (size_t)Mm * Dd * 2;      // [M,CS] bf16
constexpr size_t WS_RAW   = WS_GXR   + (size_t)Mm * CS * 2;      // [M,D] f32
constexpr size_t WS_RS    = WS_RAW   + (size_t)Mm * Dd * 4;      // RS_COUNT x [M] f32
constexpr size_t WS_COL   = WS_RS    + (size_t)RS_COUNT * Mm * 4;// [N] f32
constexpr size_t WS_SC    = WS_COL   + (size_t)Nn * 4;           // SC_COUNT f32

constexpr size_t ZERO_FLOATS = (size_t)Mm * Dd + (size_t)RS_COUNT * Mm + (size_t)Nn + SC_COUNT;

// ---------------- output offsets (reference tuple order, flattened) ----------------
constexpr size_t O_OUT     = 0;                       // [B,S,D]
constexpr size_t O_ACT     = (size_t)Mm * Dd;         // act/N        [B,S,1]
constexpr size_t O_GMAX    = O_ACT + Mm;              // gmax         [B,S,1]
constexpr size_t O_LB      = O_GMAX + Mm;             // score_lb     scalar
constexpr size_t O_STD     = O_LB + 1;                // score_std    scalar
constexpr size_t O_ES      = O_LB + 2;                // es_out       scalar
constexpr size_t O_ACTMEAN = O_LB + 3;                // active_n_mean scalar
constexpr size_t O_STRONG  = O_LB + 4;                // strong/N     [B,S,1]
constexpr size_t O_PHIB    = O_STRONG + Mm;           // phib/N       [B,S,1]
constexpr size_t O_ZMEAN   = O_PHIB + Mm;             // z_mean_active [B,S,1]
constexpr size_t O_TAUM    = O_ZMEAN + Mm;            // tau mean     scalar
constexpr size_t O_Z075    = O_TAUM + 1;
constexpr size_t O_Z030    = O_TAUM + 2;
constexpr size_t O_SKEW    = O_TAUM + 3;
constexpr size_t O_ACTSTD  = O_TAUM + 4;
constexpr size_t O_ENT     = O_TAUM + 5;

// ---------------- WMMA types ----------------
typedef __attribute__((ext_vector_type(16))) __bf16 v16bf;
typedef __attribute__((ext_vector_type(8)))  float  v8f;

__device__ inline v8f wmma_bf16(v16bf a, v16bf b, v8f c) {
  // (neg_a, A, neg_b, B, c_mod, C, reuse_a, reuse_b)
  return __builtin_amdgcn_wmma_f32_16x16x32_bf16(false, a, false, b, (short)0, c, false, false);
}

// A fragment (16x32 bf16): lanes 0-15 rows, K = kb..kb+15 contiguous, kb=16*(lane>=16)
__device__ inline v16bf frag_A(const __bf16* lds, int rbase, int lane) {
  const __bf16* p = lds + (size_t)(rbase + (lane & 15)) * LDK + ((lane & 16) ? 16 : 0);
  v16bf f;
#pragma unroll
  for (int t = 0; t < 8; ++t) { f[t] = p[t]; f[8 + t] = p[8 + t]; }
  return f;
}

// B fragment (32x16 bf16) from LDS holding the *other matrix rows* [n][K] row-major:
// lane n = lane&15, K = kb..kb+15 contiguous, kb = 16*(lane>=16)
__device__ inline v16bf frag_Brow(const __bf16* lds, int nbase, int lane) {
  const __bf16* p = lds + (size_t)(nbase + (lane & 15)) * LDK + ((lane & 16) ? 16 : 0);
  v16bf f;
#pragma unroll
  for (int t = 0; t < 16; ++t) f[t] = p[t];
  return f;
}

// B fragment from LDS in [k][n] layout (stride ld)
__device__ inline v16bf frag_Bkn(const __bf16* lds, int nbase, int ld, int lane) {
  int n  = nbase + (lane & 15);
  int kb = (lane & 16) ? 16 : 0;
  v16bf f;
#pragma unroll
  for (int t = 0; t < 16; ++t) f[t] = lds[(size_t)(kb + t) * ld + n];
  return f;
}

// ---- CDNA5 async global->LDS DMA (ASYNCcnt-tracked, no staging VGPRs) ----
__device__ inline uint32_t lds_off32(const void* p) {
  // flat LDS aperture: addr[31:0] is the LDS byte offset (ISA 10.2 aperture rules)
  return (uint32_t)(uintptr_t)p;
}
__device__ inline void async_b128(uint32_t ldsa, const void* gsrc) {
  asm volatile("global_load_async_to_lds_b128 %0, %1, off"
               :: "v"(ldsa), "v"(gsrc)
               : "memory");
}
__device__ inline void wait_async0() {
  asm volatile("s_wait_asynccnt 0x0" ::: "memory");
}

// [ROWS x 32] bf16 tile -> LDS (LDK stride), CHUNKS = ROWS/64 b128 issues per thread
template <int CHUNKS>
__device__ inline void async_tileK32(__bf16* __restrict__ dst, const __bf16* __restrict__ src,
                                     int srcld, int tid) {
#pragma unroll
  for (int i = 0; i < CHUNKS; ++i) {
    int c = tid + 256 * i;
    int row = c >> 2, kg = (c & 3) << 3;
    async_b128(lds_off32(dst + (size_t)row * LDK + kg), src + (size_t)row * srcld + kg);
  }
}
// [32 x 128] bf16 tile (write-matrix slice) -> LDS (LDW stride)
__device__ inline void async_wtile(__bf16* __restrict__ dst, const __bf16* __restrict__ src,
                                   int tid) {
#pragma unroll
  for (int i = 0; i < 2; ++i) {
    int c = tid + 256 * i;
    int row = c >> 4, ng = (c & 15) << 3;
    async_b128(lds_off32(dst + (size_t)row * LDW + ng), src + (size_t)row * Dd + ng);
  }
}

// shuffle-reduce per-row partials (vals[i][v] summed over col subtiles) into global row array
__device__ inline void row_reduce_add(const float (*vals)[8], float* red, float* gdst,
                                      int mbase, int wr, int lane, int tid) {
  __syncthreads();
  if (tid < 128) red[tid] = 0.f;
  __syncthreads();
#pragma unroll
  for (int i = 0; i < 2; ++i)
#pragma unroll
    for (int v = 0; v < 8; ++v) {
      float val = vals[i][v];
      val += __shfl_xor(val, 1); val += __shfl_xor(val, 2);
      val += __shfl_xor(val, 4); val += __shfl_xor(val, 8);
      if ((lane & 15) == 0)
        atomicAdd(&red[wr * 32 + i * 16 + v + ((lane & 16) ? 8 : 0)], val);
    }
  __syncthreads();
  if (tid < 128) atomicAdd(&gdst[mbase + tid], red[tid]);
}

__device__ inline void breduce_add(float v, float* gdst, float* sm8) {
  __syncthreads();
  v += __shfl_xor(v, 1); v += __shfl_xor(v, 2); v += __shfl_xor(v, 4);
  v += __shfl_xor(v, 8); v += __shfl_xor(v, 16);
  if ((threadIdx.x & 31) == 0) sm8[threadIdx.x >> 5] = v;
  __syncthreads();
  if (threadIdx.x == 0) {
    float s = 0.f;
    for (int i = 0; i < 8; ++i) s += sm8[i];
    atomicAdd(gdst, s);
  }
}

// ---------------- kernels ----------------

__global__ __launch_bounds__(256) void k_zero(float* p, size_t n) {
  size_t i = (size_t)blockIdx.x * 256 + threadIdx.x;
  for (size_t idx = i; idx < n; idx += (size_t)gridDim.x * 256) p[idx] = 0.f;
}

// row-normalize emb / w_read / w_write -> bf16 (blockIdx.y selects matrix)
__global__ __launch_bounds__(256) void k_norm(const float* __restrict__ emb,
                                              const float* __restrict__ wrd,
                                              const float* __restrict__ wwr,
                                              __bf16* embn, __bf16* readn, __bf16* writen) {
  __shared__ float sred[8];
  int row = blockIdx.x, which = blockIdx.y;
  const float* src = which == 0 ? emb : (which == 1 ? wrd : wwr);
  __bf16*      dst = which == 0 ? embn : (which == 1 ? readn : writen);
  const float* p = src + (size_t)row * Dd;
  float ss = 0.f;
  for (int i = threadIdx.x; i < Dd; i += 256) { float v = p[i]; ss += v * v; }
  ss += __shfl_xor(ss, 1); ss += __shfl_xor(ss, 2); ss += __shfl_xor(ss, 4);
  ss += __shfl_xor(ss, 8); ss += __shfl_xor(ss, 16);
  if ((threadIdx.x & 31) == 0) sred[threadIdx.x >> 5] = ss;
  __syncthreads();
  float tot = 0.f;
  for (int i = 0; i < 8; ++i) tot += sred[i];
  float inv = 1.f / (sqrtf(tot) + 1e-8f);
  __bf16* q = dst + (size_t)row * Dd;
  for (int i = threadIdx.x; i < Dd; i += 256) q[i] = (__bf16)(p[i] * inv);
}

__global__ __launch_bounds__(256) void k_cvt(const float* __restrict__ h,
                                             const float* __restrict__ x,
                                             __bf16* hbf, __bf16* xbf) {
  size_t i = (size_t)blockIdx.x * 256 + threadIdx.x;
  if (i < (size_t)Mm * Dd) { hbf[i] = (__bf16)h[i]; xbf[i] = (__bf16)x[i]; }
}

// ---- pass 1: score GEMM + stats (s, s^2, s^3 per row; column sums) ----
__global__ __launch_bounds__(256)
__attribute__((amdgpu_waves_per_eu(1, 2)))
void k_pass1(const __bf16* __restrict__ hbf,
             const __bf16* __restrict__ embn,
             float* __restrict__ rs,
             float* __restrict__ colsum) {
  __shared__ __bf16 As[2][128 * LDK];
  __shared__ __bf16 Bs[2][128 * LDK];
  __shared__ float  red[128];
  int tid = threadIdx.x, lane = tid & 31, wid = tid >> 5;
  int wr = wid & 3, wc = wid >> 2;
  int mbase = blockIdx.y * 128, nbase = blockIdx.x * 128;

  v8f vz = {0.f, 0.f, 0.f, 0.f, 0.f, 0.f, 0.f, 0.f};
  v8f acc[2][4];
#pragma unroll
  for (int i = 0; i < 2; ++i)
#pragma unroll
    for (int j = 0; j < 4; ++j) acc[i][j] = vz;

  const __bf16* Aseg = hbf + (size_t)mbase * Dd;
  const __bf16* Bseg = embn + (size_t)nbase * Dd;

  // prologue: async DMA first K-tile into buf0
  async_tileK32<2>(As[0], Aseg, Dd, tid);
  async_tileK32<2>(Bs[0], Bseg, Dd, tid);

  for (int k0 = 0; k0 < Dd; k0 += 32) {
    int cur = (k0 >> 5) & 1;
    bool more = (k0 + 32) < Dd;
    wait_async0();                         // own buf[cur] DMA writes complete
    __syncthreads();                       // all waves' buf[cur] published; buf[nxt] readers drained
    if (more) {                            // async DMA next tile; a full iteration to land
      async_tileK32<2>(As[cur ^ 1], Aseg + k0 + 32, Dd, tid);
      async_tileK32<2>(Bs[cur ^ 1], Bseg + k0 + 32, Dd, tid);
    }
    v16bf a0 = frag_A(As[cur], wr * 32, lane);
    v16bf a1 = frag_A(As[cur], wr * 32 + 16, lane);
#pragma unroll
    for (int j = 0; j < 4; ++j) {
      v16bf b = frag_Brow(Bs[cur], wc * 64 + j * 16, lane);
      acc[0][j] = wmma_bf16(a0, b, acc[0][j]);
      acc[1][j] = wmma_bf16(a1, b, acc[1][j]);
    }
  }

  // per-row sum / sum^2 / sum^3 partials (summed over this block's 128 columns)
  float v1[2][8], v2[2][8], v3[2][8];
#pragma unroll
  for (int i = 0; i < 2; ++i)
#pragma unroll
    for (int v = 0; v < 8; ++v) {
      float s = 0.f, s2 = 0.f, s3 = 0.f;
#pragma unroll
      for (int j = 0; j < 4; ++j) {
        float xv = acc[i][j][v];
        s += xv; s2 += xv * xv; s3 += xv * xv * xv;
      }
      v1[i][v] = s; v2[i][v] = s2; v3[i][v] = s3;
    }
  row_reduce_add(v1, red, rs + (size_t)RS_SSUM * Mm, mbase, wr, lane, tid);
  row_reduce_add(v2, red, rs + (size_t)RS_SQ * Mm, mbase, wr, lane, tid);
  row_reduce_add(v3, red, rs + (size_t)RS_CUBE * Mm, mbase, wr, lane, tid);

  // per-column partial sums (over this block's 128 rows)
  __syncthreads();
  if (tid < 128) red[tid] = 0.f;
  __syncthreads();
#pragma unroll
  for (int j = 0; j < 4; ++j) {
    float cacc = 0.f;
#pragma unroll
    for (int i = 0; i < 2; ++i)
#pragma unroll
      for (int v = 0; v < 8; ++v) cacc += acc[i][j][v];
    cacc += __shfl_xor(cacc, 16);
    if (lane < 16) atomicAdd(&red[wc * 64 + j * 16 + lane], cacc);
  }
  __syncthreads();
  if (tid < 128) atomicAdd(&colsum[nbase + tid], red[tid]);
}

// ---- pass-1 finalize: tau/std/mean per token + skew/std/tau scalar sums ----
__global__ __launch_bounds__(256) void k_rowfinal(const float* __restrict__ tau_off,
                                                  float* __restrict__ rs,
                                                  float* __restrict__ sc) {
  __shared__ float sm8[8];
  int m = blockIdx.x * 256 + threadIdx.x;
  float ssum = rs[(size_t)RS_SSUM * Mm + m];
  float ssq  = rs[(size_t)RS_SQ * Mm + m];
  float scu  = rs[(size_t)RS_CUBE * Mm + m];
  float mean = ssum / (float)Nn;
  float var  = ssq / (float)Nn - mean * mean;
  float sstd = sqrtf(fmaxf(var, 0.f)) + 1e-8f;
  float tv   = mean + tau_off[m] * sstd;
  rs[(size_t)RS_MEAN * Mm + m] = mean;
  rs[(size_t)RS_STD * Mm + m]  = sstd;
  rs[(size_t)RS_TAU * Mm + m]  = tv;
  float cmean = scu / (float)Nn;
  float c3 = cmean - 3.f * mean * sstd * sstd - mean * mean * mean;
  breduce_add(c3 / (sstd * sstd * sstd + 1e-8f), &sc[SC_SKEW], sm8);
  breduce_add(sstd, &sc[SC_STDSUM], sm8);
  breduce_add(tv, &sc[SC_TAU], sm8);
}

__global__ __launch_bounds__(256) void k_colfinal(const float* __restrict__ colsum,
                                                  float* __restrict__ sc) {
  __shared__ float sm8[8];
  int i = blockIdx.x * 256 + threadIdx.x;
  float cm = colsum[i] / (float)Mm;
  breduce_add(cm, &sc[SC_NSSUM], sm8);
  breduce_add(cm * cm, &sc[SC_NSSQ], sm8);
}

// ---- pass 2a: gate = erf-threshold(h@emb^T), xr = x@read^T, emit gate*xr (bf16) + row stats ----
__global__ __launch_bounds__(256)
__attribute__((amdgpu_waves_per_eu(1, 2)))
void k_p2a(const __bf16* __restrict__ hbf,
           const __bf16* __restrict__ xbf,
           const __bf16* __restrict__ embn,
           const __bf16* __restrict__ readn,
           float* __restrict__ rs,
           __bf16* __restrict__ gxr, int ci) {
  __shared__ __bf16 As[2][128 * LDK];
  __shared__ __bf16 Bs[2][64 * LDK];
  __shared__ float  red[128];
  __shared__ float  tauS[128], stdS[128];
  int tid = threadIdx.x, lane = tid & 31, wid = tid >> 5;
  int wr = wid & 3, wc = wid >> 2;
  int mbase = blockIdx.y * 128;
  int nb    = blockIdx.x * 64;                   // column base within chunk
  size_t grow = (size_t)ci * CS + nb;            // global neuron row base

  if (tid < 128) {
    tauS[tid] = rs[(size_t)RS_TAU * Mm + mbase + tid];
    stdS[tid] = rs[(size_t)RS_STD * Mm + mbase + tid];
  }

  v8f vzv = {0.f, 0.f, 0.f, 0.f, 0.f, 0.f, 0.f, 0.f};
  v8f accS[2][2], accX[2][2];
#pragma unroll
  for (int i = 0; i < 2; ++i)
#pragma unroll
    for (int j = 0; j < 2; ++j) { accS[i][j] = vzv; accX[i][j] = vzv; }

  for (int phase = 0; phase < 2; ++phase) {
    const __bf16* Aseg = (phase ? xbf : hbf) + (size_t)mbase * Dd;
    const __bf16* Bseg = (phase ? readn : embn) + grow * Dd;
    v8f (*acc)[2] = phase ? accX : accS;

    // prologue: async DMA first tile into buf0 (prior readers of buf0 already
    // drained by the last barrier of the previous phase)
    async_tileK32<2>(As[0], Aseg, Dd, tid);
    async_tileK32<1>(Bs[0], Bseg, Dd, tid);

    for (int k0 = 0; k0 < Dd; k0 += 32) {
      int cur = (k0 >> 5) & 1;
      bool more = (k0 + 32) < Dd;
      wait_async0();
      __syncthreads();
      if (more) {
        async_tileK32<2>(As[cur ^ 1], Aseg + k0 + 32, Dd, tid);
        async_tileK32<1>(Bs[cur ^ 1], Bseg + k0 + 32, Dd, tid);
      }
      v16bf a0 = frag_A(As[cur], wr * 32, lane);
      v16bf a1 = frag_A(As[cur], wr * 32 + 16, lane);
#pragma unroll
      for (int j = 0; j < 2; ++j) {
        v16bf b = frag_Brow(Bs[cur], wc * 32 + j * 16, lane);
        acc[0][j] = wmma_bf16(a0, b, acc[0][j]);
        acc[1][j] = wmma_bf16(a1, b, acc[1][j]);
      }
    }
  }

  // epilogue: z, phi, gate; store gate*xr (bf16)
  float z_[2][2][8], ph_[2][2][8], g_[2][2][8];
#pragma unroll
  for (int i = 0; i < 2; ++i)
#pragma unroll
    for (int j = 0; j < 2; ++j)
#pragma unroll
      for (int v = 0; v < 8; ++v) {
        int r = wr * 32 + i * 16 + v + ((lane & 16) ? 8 : 0);
        float scv = accS[i][j][v];
        float zz = (scv - tauS[r]) / stdS[r];
        float ph = 0.5f * (1.f + erff(zz * 0.7071067811865476f));
        float g  = zz > 0.f ? zz * ph : 0.f;
        z_[i][j][v] = zz; ph_[i][j][v] = ph; g_[i][j][v] = g;
        float gb = (float)(__bf16)g;
        float xr = (float)(__bf16)accX[i][j][v];
        int col = nb + wc * 32 + j * 16 + (lane & 15);
        gxr[(size_t)(mbase + r) * CS + col] = (__bf16)(gb * xr);
      }

#define ROWSTAT(EXPR, IDX)                                                     \
  do {                                                                         \
    float tv_[2][8];                                                           \
    for (int i = 0; i < 2; ++i)                                                \
      for (int v = 0; v < 8; ++v) {                                            \
        float s = 0.f;                                                         \
        for (int j = 0; j < 2; ++j) {                                          \
          float zz = z_[i][j][v]; float ph = ph_[i][j][v]; float g = g_[i][j][v]; \
          (void)zz; (void)ph; (void)g;                                         \
          s += (EXPR);                                                         \
        }                                                                      \
        tv_[i][v] = s;                                                         \
      }                                                                        \
    row_reduce_add(tv_, red, rs + (size_t)(IDX) * Mm, mbase, wr, lane, tid);   \
  } while (0)

  ROWSTAT(g, RS_WCOST);
  ROWSTAT(g > 0.f ? 1.f : 0.f, RS_ACT);
  ROWSTAT(g > 0.5f ? 1.f : 0.f, RS_STRONG);
  ROWSTAT((ph > 0.95f || ph < 0.05f) ? 1.f : 0.f, RS_PHIB);
  ROWSTAT(zz > 0.f ? zz : 0.f, RS_ZSUM);
  ROWSTAT((zz > 0.f && zz < 0.75f) ? 1.f : 0.f, RS_Z075);
  ROWSTAT((zz > 0.f && zz < 0.3f) ? 1.f : 0.f, RS_Z030);
  ROWSTAT(g * logf(g + 1e-8f), RS_GLG);
#undef ROWSTAT

  // gmax (gates are >= 0 -> int-ordered atomicMax on float bits)
  __syncthreads();
  if (tid < 128) red[tid] = 0.f;
  __syncthreads();
#pragma unroll
  for (int i = 0; i < 2; ++i)
#pragma unroll
    for (int v = 0; v < 8; ++v) {
      float mx = fmaxf(g_[i][0][v], g_[i][1][v]);
      mx = fmaxf(mx, __shfl_xor(mx, 1)); mx = fmaxf(mx, __shfl_xor(mx, 2));
      mx = fmaxf(mx, __shfl_xor(mx, 4)); mx = fmaxf(mx, __shfl_xor(mx, 8));
      if ((lane & 15) == 0)
        atomicMax((int*)&red[wr * 32 + i * 16 + v + ((lane & 16) ? 8 : 0)],
                  __float_as_int(mx));
    }
  __syncthreads();
  if (tid < 128)
    atomicMax((int*)&rs[(size_t)RS_GMAX * Mm + mbase + tid], __float_as_int(red[tid]));
}

// ---- pass 2b: raw_out += (gate*xr)[M,CS] @ write[chunk,D] ----
__global__ __launch_bounds__(256)
__attribute__((amdgpu_waves_per_eu(1, 2)))
void k_p2b(const __bf16* __restrict__ gxr,
           const __bf16* __restrict__ writen,
           float* __restrict__ raw, int ci) {
  __shared__ __bf16 Ag[2][128 * LDK];
  __shared__ __bf16 Bw[2][32 * LDW];
  int tid = threadIdx.x, lane = tid & 31, wid = tid >> 5;
  int wr = wid & 3, wc = wid >> 2;
  int mbase = blockIdx.y * 128, dbase = blockIdx.x * 128;

  v8f vzv = {0.f, 0.f, 0.f, 0.f, 0.f, 0.f, 0.f, 0.f};
  v8f acc[2][4];
#pragma unroll
  for (int i = 0; i < 2; ++i)
#pragma unroll
    for (int j = 0; j < 4; ++j) acc[i][j] = vzv;

  const __bf16* Aseg = gxr + (size_t)mbase * CS;
  const __bf16* Bseg = writen + (size_t)ci * CS * Dd + dbase;

  async_tileK32<2>(Ag[0], Aseg, CS, tid);
  async_wtile(Bw[0], Bseg, tid);

  for (int k0 = 0; k0 < CS; k0 += 32) {
    int cur = (k0 >> 5) & 1;
    bool more = (k0 + 32) < CS;
    wait_async0();
    __syncthreads();
    if (more) {
      async_tileK32<2>(Ag[cur ^ 1], Aseg + k0 + 32, CS, tid);
      async_wtile(Bw[cur ^ 1], Bseg + (size_t)(k0 + 32) * Dd, tid);
    }
    v16bf a0 = frag_A(Ag[cur], wr * 32, lane);
    v16bf a1 = frag_A(Ag[cur], wr * 32 + 16, lane);
#pragma unroll
    for (int j = 0; j < 4; ++j) {
      v16bf b = frag_Bkn(Bw[cur], wc * 64 + j * 16, LDW, lane);
      acc[0][j] = wmma_bf16(a0, b, acc[0][j]);
      acc[1][j] = wmma_bf16(a1, b, acc[1][j]);
    }
  }

  // exclusive owner per (row,col) this launch; chunk launches serialize on stream
#pragma unroll
  for (int i = 0; i < 2; ++i)
#pragma unroll
    for (int j = 0; j < 4; ++j)
#pragma unroll
      for (int v = 0; v < 8; ++v) {
        int r = mbase + wr * 32 + i * 16 + v + ((lane & 16) ? 8 : 0);
        int c = dbase + wc * 64 + j * 16 + (lane & 15);
        size_t off = (size_t)r * Dd + c;
        raw[off] += acc[i][j][v];
      }
}

// ---- finals ----
__global__ __launch_bounds__(256) void k_out(const float* __restrict__ raw,
                                             const float* __restrict__ rs,
                                             float* __restrict__ out) {
  size_t base = ((size_t)blockIdx.x * 256 + threadIdx.x) * 4;
  int m = (int)(base / Dd);
  float den = fmaxf(rs[(size_t)RS_WCOST * Mm + m], 1.f);
#pragma unroll
  for (int t = 0; t < 4; ++t)
    out[O_OUT + base + t] = (float)(__bf16)(raw[base + t] / den);  // bf16 round-trip
}

__global__ __launch_bounds__(256) void k_rowout(const float* __restrict__ rs,
                                                float* __restrict__ sc,
                                                float* __restrict__ out) {
  __shared__ float sm8[8];
  int m = blockIdx.x * 256 + threadIdx.x;
  float act = rs[(size_t)RS_ACT * Mm + m];
  float wc  = rs[(size_t)RS_WCOST * Mm + m];
  float st  = rs[(size_t)RS_STRONG * Mm + m];
  float pb  = rs[(size_t)RS_PHIB * Mm + m];
  float zs  = rs[(size_t)RS_ZSUM * Mm + m];
  float z75 = rs[(size_t)RS_Z075 * Mm + m];
  float z30 = rs[(size_t)RS_Z030 * Mm + m];
  float gl  = rs[(size_t)RS_GLG * Mm + m];
  float gm  = rs[(size_t)RS_GMAX * Mm + m];
  out[O_ACT + m]    = act / (float)Nn;
  out[O_GMAX + m]   = gm;
  out[O_STRONG + m] = st / (float)Nn;
  out[O_PHIB + m]   = pb / (float)Nn;
  out[O_ZMEAN + m]  = zs / (act + 1e-8f);
  float gse = wc + 1e-8f;
  breduce_add(wc, &sc[SC_ES], sm8);
  breduce_add(act, &sc[SC_ACT], sm8);
  breduce_add(act * act, &sc[SC_ACTSQ], sm8);
  breduce_add(z75 / (act + 1e-8f), &sc[SC_Z075], sm8);
  breduce_add(z30 / (act + 1e-8f), &sc[SC_Z030], sm8);
  breduce_add(-gl / gse + logf(gse), &sc[SC_ENT], sm8);
}

__global__ void k_scalars(const float* __restrict__ sc, float* __restrict__ out) {
  if (threadIdx.x == 0 && blockIdx.x == 0) {
    float ms = sc[SC_NSSUM] / (float)Nn;
    float vs = sc[SC_NSSQ] / (float)Nn - ms * ms;
    out[O_LB]      = vs / (ms * ms + vs + 0.01f);
    out[O_STD]     = sc[SC_STDSUM] / (float)Mm;
    out[O_ES]      = sc[SC_ES] / (float)Mm;
    out[O_ACTMEAN] = sc[SC_ACT] / (float)Mm;
    out[O_TAUM]    = sc[SC_TAU] / (float)Mm;
    out[O_Z075]    = sc[SC_Z075] / (float)Mm;
    out[O_Z030]    = sc[SC_Z030] / (float)Mm;
    out[O_SKEW]    = sc[SC_SKEW] / (float)Mm;
    float am = sc[SC_ACT] / (float)Mm;
    out[O_ACTSTD] = sqrtf(fmaxf(sc[SC_ACTSQ] / (float)Mm - am * am, 0.f));
    out[O_ENT]    = sc[SC_ENT] / (float)Mm;
  }
}

// ---------------- host entry ----------------
extern "C" void kernel_launch(void* const* d_in, const int* in_sizes, int n_in,
                              void* d_out, int out_size, void* d_ws, size_t ws_size,
                              hipStream_t stream) {
  (void)in_sizes; (void)n_in; (void)out_size; (void)ws_size;
  const float* x       = (const float*)d_in[0];
  const float* h       = (const float*)d_in[1];
  const float* emb     = (const float*)d_in[2];
  const float* tau_off = (const float*)d_in[3];
  const float* wread   = (const float*)d_in[4];
  const float* wwrite  = (const float*)d_in[5];
  // d_in[6] = n_chunks (compile-time NCH here)

  char* ws = (char*)d_ws;
  __bf16* embn   = (__bf16*)(ws + WS_EMB);
  __bf16* readn  = (__bf16*)(ws + WS_READ);
  __bf16* writen = (__bf16*)(ws + WS_WRITE);
  __bf16* hbf    = (__bf16*)(ws + WS_HBF);
  __bf16* xbf    = (__bf16*)(ws + WS_XBF);
  __bf16* gxr    = (__bf16*)(ws + WS_GXR);
  float*  raw    = (float*)(ws + WS_RAW);
  float*  rsarr  = (float*)(ws + WS_RS);
  float*  colsum = (float*)(ws + WS_COL);
  float*  scarr  = (float*)(ws + WS_SC);
  float*  out    = (float*)d_out;

  // zero raw_out + all stat accumulators (contiguous region starting at WS_RAW)
  k_zero<<<4096, 256, 0, stream>>>(raw, ZERO_FLOATS);
  // normalize three weight matrices -> bf16; convert h,x -> bf16
  k_norm<<<dim3(Nn, 3), 256, 0, stream>>>(emb, wread, wwrite, embn, readn, writen);
  k_cvt<<<(Mm * Dd) / 256, 256, 0, stream>>>(h, x, hbf, xbf);
  // pass 1: score statistics
  k_pass1<<<dim3(Nn / 128, Mm / 128), 256, 0, stream>>>(hbf, embn, rsarr, colsum);
  k_rowfinal<<<Mm / 256, 256, 0, stream>>>(tau_off, rsarr, scarr);
  k_colfinal<<<Nn / 256, 256, 0, stream>>>(colsum, scarr);
  // pass 2: per-chunk gate/xr then out-GEMM accumulate
  for (int ci = 0; ci < NCH; ++ci) {
    k_p2a<<<dim3(CS / 64, Mm / 128), 256, 0, stream>>>(hbf, xbf, embn, readn, rsarr, gxr, ci);
    k_p2b<<<dim3(Dd / 128, Mm / 128), 256, 0, stream>>>(gxr, writen, raw, ci);
  }
  // finals
  k_out<<<(Mm * Dd) / 1024, 256, 0, stream>>>(raw, rsarr, out);
  k_rowout<<<Mm / 256, 256, 0, stream>>>(rsarr, scarr, out);
  k_scalars<<<1, 64, 0, stream>>>(scarr, out);
}
